// WaymoMultiModalMamba_9491877724670
// MI455X (gfx1250) — matrix operand order repaired and checked by reading, not compiled, not run on hardware
//
#include <hip/hip_runtime.h>
#include <hip/hip_bf16.h>
#include <math.h>

// ---------------------------------------------------------------------------
// Model dims
// ---------------------------------------------------------------------------
#define BB   32
#define SS   256
#define BS   (BB*SS)      // 8192 rows
#define DD   128
#define II   256
#define NN   16
#define RR   8
#define KK   4
#define LL   4
#define ZZ   128
#define FUT  20
#define KV   8192         // vision K = 8*1024
#define H1D  512

typedef _Float16 h8v  __attribute__((ext_vector_type(8)));
typedef _Float16 v16h __attribute__((ext_vector_type(16)));
typedef float    v8f  __attribute__((ext_vector_type(8)));

__device__ __forceinline__ float gelu_exact(float v) {
    return 0.5f * v * (1.0f + erff(v * 0.70710678118654752f));
}
__device__ __forceinline__ float silu(float v) {
    return v / (1.0f + expf(-v));
}

// ---------------------------------------------------------------------------
// Generic WMMA GEMM: C(MxN) = epilogue(A(MxK,f32) * B(KxN,f32))
// Block tile 128x128, 8 waves (4 in M x 2 in N), each wave 32x64 via
// 2x4 v_wmma_f32_16x16x32_f16.  f32 operands are converted to f16 while
// staging into LDS (no separate conversion pass over the 268MB x_cam; A-panel
// re-reads across the N-tile passes hit the 192MB L2).
// Software pipelined: next K-tile's global loads are issued before the current
// tile's fragment loads + WMMAs, so s_wait_loadcnt lands after the MMA work.
// Requires: M%128==0, N%128==0, K%32==0, lda%4==0, ldb%4==0 (all uses satisfy).
// MODE 0: out = gelu(acc + bias[col]);  MODE 1: out = acc;  MODE 2: out += acc
// ---------------------------------------------------------------------------
template<int MODE>
__global__ __launch_bounds__(256)
void wmma_gemm(const float* __restrict__ A, const float* __restrict__ Bw,
               const float* __restrict__ bias, float* __restrict__ C,
               int M, int N, int Kd, int lda, int ldb, int ldc)
{
    __shared__ __align__(16) _Float16 lA[128 * 32];   // 8 KB
    __shared__ __align__(16) _Float16 lB[32 * 128];   // 8 KB

    const int tid  = threadIdx.x;
    const int lane = tid & 31;
    const int wave = tid >> 5;
    const int wm   = wave >> 1;      // 0..3  (M direction)
    const int wn   = wave & 1;       // 0..1  (N direction)
    const int bm0  = blockIdx.y * 128;
    const int bn0  = blockIdx.x * 128;
    const int g    = lane >> 4;      // half-wave group
    const int l15  = lane & 15;

    // staging thread mapping: A 128x32 (4 float4/thread), B 32x128 (4 float4/thread)
    const int arow = tid >> 3;          // 0..31, step 32 per iter
    const int ac4  = (tid & 7) * 4;     // 0..28
    const int brow = tid >> 5;          // 0..7,  step 8 per iter
    const int bc4  = (tid & 31) * 4;    // 0..124

    float4 ra[4], rb[4];

    auto issue_loads = [&](int k0) {
#pragma unroll
        for (int it = 0; it < 4; ++it)
            ra[it] = *(const float4*)(A + (size_t)(bm0 + arow + it * 32) * lda + k0 + ac4);
#pragma unroll
        for (int it = 0; it < 4; ++it)
            rb[it] = *(const float4*)(Bw + (size_t)(k0 + brow + it * 8) * ldb + bn0 + bc4);
    };
    auto store_lds = [&]() {
#pragma unroll
        for (int it = 0; it < 4; ++it) {
            _Float16* d = &lA[(arow + it * 32) * 32 + ac4];
            d[0] = (_Float16)ra[it].x; d[1] = (_Float16)ra[it].y;
            d[2] = (_Float16)ra[it].z; d[3] = (_Float16)ra[it].w;
        }
#pragma unroll
        for (int it = 0; it < 4; ++it) {
            _Float16* d = &lB[(brow + it * 8) * 128 + bc4];
            d[0] = (_Float16)rb[it].x; d[1] = (_Float16)rb[it].y;
            d[2] = (_Float16)rb[it].z; d[3] = (_Float16)rb[it].w;
        }
    };

    v8f c[2][4] = {};

    issue_loads(0);
    store_lds();
    __syncthreads();

    for (int k0 = 0; k0 < Kd; k0 += 32) {
        const bool more = (k0 + 32 < Kd);
        if (more) issue_loads(k0 + 32);          // overlap with MMA below

        // ---- fragments per ISA 7.12.2 layouts ----
        v16h af[2], bf[4];
#pragma unroll
        for (int ar = 0; ar < 2; ++ar) {
            const int row = wm * 32 + ar * 16 + l15;           // M within tile
            h8v lo = *(const h8v*)&lA[row * 32 + g * 8];       // K = g*8 .. +7
            h8v hi = *(const h8v*)&lA[row * 32 + 16 + g * 8];  // K = 16+g*8 .. +7
            af[ar] = __builtin_shufflevector(lo, hi, 0,1,2,3,4,5,6,7,8,9,10,11,12,13,14,15);
        }
#pragma unroll
        for (int bc = 0; bc < 4; ++bc) {
            const int colb = wn * 64 + bc * 16 + l15;          // lane <-> N col, K = g*8.., 16+g*8..
#pragma unroll
            for (int s = 0; s < 16; ++s) {
                const int kk = (s < 8) ? (g * 8 + s) : (16 + g * 8 + (s - 8));
                bf[bc][s] = lB[kk * 128 + colb];
            }
        }

#pragma unroll
        for (int ar = 0; ar < 2; ++ar)
#pragma unroll
            for (int bc = 0; bc < 4; ++bc)
                c[ar][bc] = __builtin_amdgcn_wmma_f32_16x16x32_f16(
                    false, af[ar], false, bf[bc], (short)0, c[ar][bc], false, false);

        if (more) {
            __syncthreads();     // all waves done reading LDS
            store_lds();         // waits loads here, after the MMA work
            __syncthreads();     // LDS tile ready
        }
    }

    // ---- epilogue: lane layout  M = r + 8*g,  N = l15 ----
#pragma unroll
    for (int ar = 0; ar < 2; ++ar) {
#pragma unroll
        for (int bc = 0; bc < 4; ++bc) {
            const int colg = bn0 + wn * 64 + bc * 16 + l15;
#pragma unroll
            for (int r = 0; r < 8; ++r) {
                const int rowg = bm0 + wm * 32 + ar * 16 + g * 8 + r;
                float v = c[ar][bc][r];
                float* dst = C + (size_t)rowg * ldc + colg;
                if (MODE == 0)      *dst = gelu_exact(v + bias[colg]);
                else if (MODE == 1) *dst = v;
                else                *dst = *dst + v;
            }
        }
    }
}

// ---------------------------------------------------------------------------
// Combined fusion weights:  fused = gelu(H1 @ (W2@FwTop) + Xe @ (Ew@FwBot) + bc)
// Wc is 576x128 (rows 0..511 from vis_w2 path, 512..575 from ego path)
// ---------------------------------------------------------------------------
__global__ void combine_wc_kernel(const float* __restrict__ vis_w2,
                                  const float* __restrict__ ego_w,
                                  const float* __restrict__ fus_w,
                                  float* __restrict__ Wc)
{
    int gid = blockIdx.x * blockDim.x + threadIdx.x;
    if (gid >= 576 * 128) return;
    int k = gid / 128, n = gid % 128;
    float acc = 0.f;
    if (k < 512) {
        for (int j = 0; j < 128; ++j) acc += vis_w2[k * 128 + j] * fus_w[j * 128 + n];
    } else {
        int kk = k - 512;
        for (int j = 0; j < 128; ++j) acc += ego_w[kk * 128 + j] * fus_w[(128 + j) * 128 + n];
    }
    Wc[gid] = acc;
}

__global__ void combine_bc_kernel(const float* __restrict__ vis_b2,
                                  const float* __restrict__ ego_b,
                                  const float* __restrict__ fus_w,
                                  const float* __restrict__ fus_b,
                                  float* __restrict__ bc)
{
    int n = threadIdx.x;
    if (n >= 128) return;
    float acc = fus_b[n];
    for (int j = 0; j < 128; ++j) acc += vis_b2[j] * fus_w[j * 128 + n];
    for (int j = 0; j < 128; ++j) acc += ego_b[j] * fus_w[(128 + j) * 128 + n];
    bc[n] = acc;
}

// copy x_ego into columns 512..575 of the A2 (8192x576) activation matrix
__global__ void ego_copy_kernel(const float* __restrict__ x_ego, float* __restrict__ A2)
{
    int gid = blockIdx.x * blockDim.x + threadIdx.x;
    if (gid >= BS * 64) return;
    int row = gid / 64, j = gid % 64;
    A2[(size_t)row * 576 + 512 + j] = x_ego[gid];
}

// ---------------------------------------------------------------------------
// RMSNorm: grid BS blocks x 128 threads
// ---------------------------------------------------------------------------
__global__ void rms_kernel(const float* __restrict__ x, const float* __restrict__ w,
                           float* __restrict__ out)
{
    __shared__ float red[128];
    const int row = blockIdx.x, t = threadIdx.x;
    float v = x[(size_t)row * 128 + t];
    red[t] = v * v;
    __syncthreads();
    for (int off = 64; off > 0; off >>= 1) {
        if (t < off) red[t] += red[t + off];
        __syncthreads();
    }
    float scale = rsqrtf(red[0] * (1.0f / 128.0f) + 1e-5f);
    out[(size_t)row * 128 + t] = v * scale * w[t];
}

// depthwise causal conv (K=4) + bias + silu; h = hg[...,:256]
__global__ void conv_silu_kernel(const float* __restrict__ hg,
                                 const float* __restrict__ cw,  // (I,K)
                                 const float* __restrict__ cb,  // (I,)
                                 float* __restrict__ hc)
{
    const int row = blockIdx.x, i = threadIdx.x;
    const int s = row & (SS - 1), b = row >> 8;
    float acc = cb[i];
#pragma unroll
    for (int k = 0; k < KK; ++k) {
        int sp = s + k - (KK - 1);
        if (sp >= 0) acc += hg[(size_t)(b * SS + sp) * 512 + i] * cw[i * KK + k];
    }
    hc[(size_t)row * II + i] = silu(acc);
}

// ssm = hc(8192x256) @ xpw(256x40)    (ragged N=40 -> plain VALU kernel)
__global__ void xproj_kernel(const float* __restrict__ hc,
                             const float* __restrict__ xpw,
                             float* __restrict__ ssm)
{
    int gid = blockIdx.x * blockDim.x + threadIdx.x;
    if (gid >= BS * 40) return;
    int row = gid / 40, n = gid % 40;
    const float* hr = hc + (size_t)row * II;
    float acc = 0.f;
    for (int j = 0; j < II; ++j) acc += hr[j] * xpw[j * 40 + n];
    ssm[gid] = acc;
}

// dt = softplus(ssm[:, :8] @ dtw(8x256) + dtb)
__global__ void dt_kernel(const float* __restrict__ ssm,
                          const float* __restrict__ dtw,
                          const float* __restrict__ dtb,
                          float* __restrict__ dto)
{
    const int row = blockIdx.x, i = threadIdx.x;
    float acc = dtb[i];
#pragma unroll
    for (int r = 0; r < RR; ++r) acc += ssm[(size_t)row * 40 + r] * dtw[r * II + i];
    dto[(size_t)row * II + i] = (acc > 20.0f) ? acc : log1pf(expf(acc));
}

// ---------------------------------------------------------------------------
// Selective scan: one block per batch b, thread i in [0,256); state (N=16)
// kept in registers; per-batch B/C (S x 32 floats) staged once into LDS.
// Fuses  y = (scan + h*D) * silu(gate)  (gate = hg[...,256+i]).
// ---------------------------------------------------------------------------
__global__ __launch_bounds__(256)
void scan_kernel(const float* __restrict__ dto, const float* __restrict__ hc,
                 const float* __restrict__ ssm, const float* __restrict__ hg,
                 const float* __restrict__ alog, const float* __restrict__ dpar,
                 float* __restrict__ y)
{
    __shared__ float lBC[SS][32];     // [s][n<16]=B, [s][16+n]=C  (32 KB)
    const int b = blockIdx.x, i = threadIdx.x;
    {
        const float* src = ssm + (size_t)(b * SS + i) * 40 + RR;
        for (int j = 0; j < 32; ++j) lBC[i][j] = src[j];
    }
    __syncthreads();

    float Aneg[NN];
#pragma unroll
    for (int n = 0; n < NN; ++n) Aneg[n] = -expf(alog[i * NN + n]);
    const float dp = dpar[i];

    float st[NN];
#pragma unroll
    for (int n = 0; n < NN; ++n) st[n] = 0.f;

    for (int s = 0; s < SS; ++s) {
        const size_t row = (size_t)(b * SS + s);
        const float dtv = dto[row * II + i];
        const float hv  = hc[row * II + i];
        float acc = 0.f;
#pragma unroll
        for (int n = 0; n < NN; ++n) {
            float dA = expf(dtv * Aneg[n]);
            st[n] = st[n] * dA + dtv * lBC[s][n] * hv;
            acc += st[n] * lBC[s][16 + n];
        }
        const float gate = hg[row * 512 + II + i];
        y[row * II + i] = (acc + hv * dp) * silu(gate);
    }
}

// ---------------------------------------------------------------------------
// Head: per-batch block; mu/lv/z + KL partial + relu-MLP -> out[b*40..]
// ---------------------------------------------------------------------------
__global__ __launch_bounds__(256)
void head_kernel(const float* __restrict__ xn, const float* __restrict__ vib_eps,
                 const float* __restrict__ mu_w, const float* __restrict__ mu_b,
                 const float* __restrict__ lv_w, const float* __restrict__ lv_b,
                 const float* __restrict__ h1_w, const float* __restrict__ h1_b,
                 const float* __restrict__ h2_w, const float* __restrict__ h2_b,
                 float* __restrict__ out, float* __restrict__ klp)
{
    __shared__ float sl[128], sz[128], sh[256], sred[256];
    const int b = blockIdx.x, t = threadIdx.x;
    const size_t lastrow = (size_t)(b * SS + (SS - 1)) * 128;
    if (t < 128) sl[t] = xn[lastrow + t];
    __syncthreads();

    float klv = 0.f;
    if (t < 128) {
        float mu = mu_b[t], lv = lv_b[t];
        for (int d = 0; d < 128; ++d) {
            float xv = sl[d];
            mu += xv * mu_w[d * ZZ + t];
            lv += xv * lv_w[d * ZZ + t];
        }
        sz[t] = mu + vib_eps[b * ZZ + t] * expf(0.5f * lv);
        klv = 1.0f + lv - mu * mu - expf(lv);
    }
    sred[t] = klv;
    __syncthreads();
    for (int off = 128; off > 0; off >>= 1) {
        if (t < off) sred[t] += sred[t + off];
        __syncthreads();
    }
    if (t == 0) klp[b] = sred[0];

    float acc1 = h1_b[t];
    for (int z = 0; z < ZZ; ++z) acc1 += sz[z] * h1_w[z * 256 + t];
    sh[t] = fmaxf(acc1, 0.f);
    __syncthreads();
    if (t < FUT * 2) {
        float acc = h2_b[t];
        for (int j = 0; j < 256; ++j) acc += sh[j] * h2_w[j * (FUT * 2) + t];
        out[b * (FUT * 2) + t] = acc;
    }
}

__global__ void kl_final_kernel(const float* __restrict__ klp, float* __restrict__ out)
{
    if (threadIdx.x == 0) {
        float s = 0.f;
        for (int b = 0; b < BB; ++b) s += klp[b];
        out[BB * FUT * 2] = -0.5f * s / (float)(BB * ZZ);
    }
}

// ---------------------------------------------------------------------------
// Launch
// ---------------------------------------------------------------------------
extern "C" void kernel_launch(void* const* d_in, const int* in_sizes, int n_in,
                              void* d_out, int out_size, void* d_ws, size_t ws_size,
                              hipStream_t stream)
{
    const float* x_cam   = (const float*)d_in[0];
    const float* x_ego   = (const float*)d_in[1];
    const float* vib_eps = (const float*)d_in[2];
    const float* vis_w1  = (const float*)d_in[3];
    const float* vis_b1  = (const float*)d_in[4];
    const float* vis_w2  = (const float*)d_in[5];
    const float* vis_b2  = (const float*)d_in[6];
    const float* ego_w   = (const float*)d_in[7];
    const float* ego_b   = (const float*)d_in[8];
    const float* fus_w   = (const float*)d_in[9];
    const float* fus_b   = (const float*)d_in[10];
    const float* norm_w  = (const float*)d_in[11];
    const float* inw     = (const float*)d_in[12];
    const float* cw      = (const float*)d_in[13];
    const float* cb      = (const float*)d_in[14];
    const float* xpw     = (const float*)d_in[15];
    const float* dtw     = (const float*)d_in[16];
    const float* dtb     = (const float*)d_in[17];
    const float* alog    = (const float*)d_in[18];
    const float* dpar    = (const float*)d_in[19];
    const float* outw    = (const float*)d_in[20];
    const float* normf_w = (const float*)d_in[21];
    const float* mu_w    = (const float*)d_in[22];
    const float* mu_b    = (const float*)d_in[23];
    const float* lv_w    = (const float*)d_in[24];
    const float* lv_b    = (const float*)d_in[25];
    const float* h1_w    = (const float*)d_in[26];
    const float* h1_b    = (const float*)d_in[27];
    const float* h2_w    = (const float*)d_in[28];
    const float* h2_b    = (const float*)d_in[29];
    float* out = (float*)d_out;

    // workspace carve-up (floats)
    float* ws   = (float*)d_ws;
    float* A2   = ws;                         // 8192 x 576
    float* Wc   = A2   + (size_t)BS * 576;    // 576 x 128
    float* bc   = Wc   + 576 * 128;           // 128
    float* xbuf = bc   + 128;                 // 8192 x 128
    float* xn   = xbuf + (size_t)BS * DD;     // 8192 x 128
    float* hg   = xn   + (size_t)BS * DD;     // 8192 x 512
    float* hcv  = hg   + (size_t)BS * 512;    // 8192 x 256
    float* ssm  = hcv  + (size_t)BS * II;     // 8192 x 40
    float* dto  = ssm  + (size_t)BS * 40;     // 8192 x 256
    float* ybuf = dto  + (size_t)BS * II;     // 8192 x 256
    float* klp  = ybuf + (size_t)BS * II;     // 32

    // 1) combined fusion weights + ego columns of A2
    combine_wc_kernel<<<(576 * 128 + 255) / 256, 256, 0, stream>>>(vis_w2, ego_w, fus_w, Wc);
    combine_bc_kernel<<<1, 128, 0, stream>>>(vis_b2, ego_b, fus_w, fus_b, bc);
    ego_copy_kernel<<<(BS * 64 + 255) / 256, 256, 0, stream>>>(x_ego, A2);

    // 2) GEMM1: H1 = gelu(x_cam @ vis_w1 + b1)  -> A2 cols [0,512)
    wmma_gemm<0><<<dim3(H1D / 128, BS / 128), 256, 0, stream>>>(
        x_cam, vis_w1, vis_b1, A2, BS, H1D, KV, KV, H1D, 576);

    // 3) fused = gelu(A2 @ Wc + bc)  (K=576, N=128) -> xbuf
    wmma_gemm<0><<<dim3(DD / 128, BS / 128), 256, 0, stream>>>(
        A2, Wc, bc, xbuf, BS, DD, 576, 576, DD, DD);

    // 4) Mamba layers
    for (int l = 0; l < LL; ++l) {
        const float* inw_l  = inw  + (size_t)l * DD * 2 * II;
        const float* cw_l   = cw   + (size_t)l * II * KK;
        const float* cb_l   = cb   + (size_t)l * II;
        const float* xpw_l  = xpw  + (size_t)l * II * 40;
        const float* dtw_l  = dtw  + (size_t)l * RR * II;
        const float* dtb_l  = dtb  + (size_t)l * II;
        const float* alog_l = alog + (size_t)l * II * NN;
        const float* dpar_l = dpar + (size_t)l * II;
        const float* outw_l = outw + (size_t)l * II * DD;
        const float* nw_l   = norm_w + (size_t)l * DD;

        rms_kernel<<<BS, 128, 0, stream>>>(xbuf, nw_l, xn);
        wmma_gemm<1><<<dim3(512 / 128, BS / 128), 256, 0, stream>>>(
            xn, inw_l, nullptr, hg, BS, 512, DD, DD, 512, 512);
        conv_silu_kernel<<<BS, II, 0, stream>>>(hg, cw_l, cb_l, hcv);
        xproj_kernel<<<(BS * 40 + 255) / 256, 256, 0, stream>>>(hcv, xpw_l, ssm);
        dt_kernel<<<BS, II, 0, stream>>>(ssm, dtw_l, dtb_l, dto);
        scan_kernel<<<BB, II, 0, stream>>>(dto, hcv, ssm, hg, alog_l, dpar_l, ybuf);
        wmma_gemm<2><<<dim3(DD / 128, BS / 128), 256, 0, stream>>>(
            ybuf, outw_l, nullptr, xbuf, BS, DD, II, II, DD, DD);
    }

    // 5) final rms + head
    rms_kernel<<<BS, 128, 0, stream>>>(xbuf, normf_w, xn);
    head_kernel<<<BB, 256, 0, stream>>>(xn, vib_eps, mu_w, mu_b, lv_w, lv_b,
                                        h1_w, h1_b, h2_w, h2_b, out, klp);
    kl_final_kernel<<<1, 32, 0, stream>>>(klp, out);
}
